// FCNSubNet_91156385890722
// MI455X (gfx1250) — matrix operand order repaired and checked by reading, not compiled, not run on hardware
//
#include <hip/hip_runtime.h>

// ---------------- CDNA5 WMMA types ----------------
typedef __attribute__((ext_vector_type(16))) __bf16 v16bf;
typedef __attribute__((ext_vector_type(8)))  float  v8f;

// ---------------- problem constants ----------------
#define Bb   4
#define Cc   256
#define HW   4096          // 64*64
#define Hh   64
#define Ww   64
#define K9   9
#define KD   2304          // Cc*9  (GEMM K dimension, k = c*9 + tap)
#define KS   2312          // LDS row stride in bf16 (pad 8 -> bank spread)
#define PX   32            // pixels per workgroup tile (N tile)
#define NPIX 16384         // Bb*HW
#define SMEM_BYTES (PX*KS*2 + 4*PX*K9*4 + PX*K9*4)   // 153,728 B

__device__ __forceinline__ unsigned short f2bf(float f) {
  unsigned u = __float_as_uint(f);
  return (unsigned short)((u + 0x7FFFu + ((u >> 16) & 1u)) >> 16); // RNE
}

// -------- weight fp32 -> bf16 row-major [M][KD] --------
__global__ __launch_bounds__(256) void cvt_w_main(const float* __restrict__ w,
                                                  unsigned short* __restrict__ o) {
  int i = blockIdx.x * 256 + threadIdx.x;
  if (i < 256 * KD) o[i] = f2bf(w[i]);
}
// 27 rows -> 64 rows zero padded
__global__ __launch_bounds__(256) void cvt_w_off(const float* __restrict__ w,
                                                 unsigned short* __restrict__ o) {
  int i = blockIdx.x * 256 + threadIdx.x;
  if (i < 64 * KD) o[i] = (i < 27 * KD) ? f2bf(w[i]) : (unsigned short)0;
}

// ============================================================
// Fused (im2col | deform-sample) + WMMA GEMM + bias (+ReLU).
// MODE 0: offset conv. fixed taps, M=64 (27 real), MT=1 NT=1,
//         wave grid 4(M) x 2(px).  out = om[B,27,H,W]
// MODE 1: main conv. offsets+mask from om, M=256, MT=2 NT=2:
//         wave owns 32 rows x 32 px -> each A frag feeds 2 WMMAs,
//         each B frag feeds 2 WMMAs (1 global b128 per WMMA).
// Block: 256 threads (8 wave32), tile = M_total x 32 pixels.
// ============================================================
template<int MODE, int MT, int NT>
__global__ __launch_bounds__(256)
void deform_gemm(const float* __restrict__ x,            // [B,256,64,64]
                 const float* __restrict__ om,           // [B,27,64,64] (MODE 1)
                 const unsigned short* __restrict__ wbf, // bf16 [M][KD]
                 const float* __restrict__ bias,
                 float* __restrict__ out)
{
  extern __shared__ char raw[];
  unsigned short* smS = (unsigned short*)raw;               // [PX][KS] bf16
  float* pw = (float*)(raw + PX * KS * 2);                  // [4][PX*K9]
  int*   pc = (int*)(pw + 4 * PX * K9);                     // [PX*K9]

  const int tid  = threadIdx.x;
  const int pix0 = blockIdx.x * PX;
  const int b    = pix0 >> 12;
  const int pos0 = pix0 & (HW - 1);
  const int h    = pos0 >> 6;
  const int w0   = pos0 & 63;

  // ---- phase 0: per-(pixel,tap) sampling params ----
  for (int i = tid; i < PX * K9; i += 256) {
    int px = i / K9, tap = i % K9;
    int w = w0 + px;
    float ys, xs, m;
    if (MODE == 1) {
      const float* omb = om + (size_t)b * 27 * HW + h * 64 + w;
      float dy = omb[(2 * tap) * HW];
      float dx = omb[(2 * tap + 1) * HW];
      float z  = omb[(18 + tap) * HW];
      m  = 1.f / (1.f + __expf(-z));
      ys = (float)(h + tap / 3 - 1) + dy;
      xs = (float)(w + tap % 3 - 1) + dx;
    } else {
      m  = 1.f;
      ys = (float)(h + tap / 3 - 1);
      xs = (float)(w + tap % 3 - 1);
    }
    float y0f = floorf(ys), x0f = floorf(xs);
    int   y0 = (int)y0f,    x0 = (int)x0f;
    float fy = ys - y0f,    fx = xs - x0f;
    int   y1 = y0 + 1,      x1 = x0 + 1;
    float vy0 = (y0 >= 0 && y0 < Hh) ? 1.f : 0.f;
    float vy1 = (y1 >= 0 && y1 < Hh) ? 1.f : 0.f;
    float vx0 = (x0 >= 0 && x0 < Ww) ? 1.f : 0.f;
    float vx1 = (x1 >= 0 && x1 < Ww) ? 1.f : 0.f;
    pw[0 * (PX*K9) + i] = m * (1.f - fy) * (1.f - fx) * vy0 * vx0;
    pw[1 * (PX*K9) + i] = m * (1.f - fy) * fx         * vy0 * vx1;
    pw[2 * (PX*K9) + i] = m * fy         * (1.f - fx) * vy1 * vx0;
    pw[3 * (PX*K9) + i] = m * fy         * fx         * vy1 * vx1;
    int y0c = min(max(y0, 0), Hh - 1), x0c = min(max(x0, 0), Ww - 1);
    int y1c = min(max(y1, 0), Hh - 1), x1c = min(max(x1, 0), Ww - 1);
    pc[i] = y0c | (x0c << 8) | (y1c << 16) | (x1c << 24);
  }
  __syncthreads();

  // ---- phase 1: fill sampled tile (bf16) into LDS ----
  const int lane = tid & 31;   // lane == pixel within tile
  const int wave = tid >> 5;
  const float* xb = x + (size_t)b * Cc * HW;
  for (int idx = wave; idx < KD; idx += 8) {   // idx = c*9 + tap == GEMM k
    int c = idx / 9, tap = idx - 9 * c;
    int pi = lane * K9 + tap;
    float a0 = pw[pi];
    int c4 = pc[pi];
    int y0 = c4 & 255, x0c = (c4 >> 8) & 255, y1 = (c4 >> 16) & 255, x1c = (c4 >> 24) & 255;
    const float* xp = xb + (size_t)c * HW;
    float v;
    if (MODE == 0) {
      v = a0 * xp[y0 * 64 + x0c];
    } else {
      float a1 = pw[(PX*K9) + pi], a2 = pw[2*(PX*K9) + pi], a3 = pw[3*(PX*K9) + pi];
      v = a0 * xp[y0 * 64 + x0c] + a1 * xp[y0 * 64 + x1c]
        + a2 * xp[y1 * 64 + x0c] + a3 * xp[y1 * 64 + x1c];
    }
    smS[lane * KS + idx] = f2bf(v);
  }
  __syncthreads();

  // ---- phase 2: WMMA GEMM  D = W(MxK) * S(KxN) ----
  const int ln16 = lane & 15;
  const bool hi  = lane >= 16;
  // wave -> (mBase, pxBase)
  const int mBase  = (MODE == 1) ? wave * (MT * 16) : (wave >> 1) * 16;
  const int pxBase = (MODE == 1) ? 0               : (wave & 1) * 16;

  v8f acc[MT][NT];
#pragma unroll
  for (int t = 0; t < MT; t++)
#pragma unroll
    for (int n = 0; n < NT; n++)
#pragma unroll
      for (int r = 0; r < 8; r++) acc[t][n][r] = 0.f;

  // B fragment sources: lane -> column px, lo lanes K 0..15, hi lanes K 16..31
  const unsigned short* srcB[NT];
#pragma unroll
  for (int n = 0; n < NT; n++)
    srcB[n] = smS + (pxBase + n * 16 + ln16) * KS + (hi ? 16 : 0);
  // A fragment source: lane -> row o, lo lanes K 0..15, hi lanes K 16..31
  const unsigned short* wrow = wbf + (size_t)(mBase + ln16) * KD + (hi ? 16 : 0);

  for (int kb = 0; kb < KD; kb += 32) {
    v16bf fB[NT];
#pragma unroll
    for (int n = 0; n < NT; n++) {
      ((uint4*)&fB[n])[0] = *(const uint4*)(srcB[n] + kb);       // 8 bf16
      ((uint4*)&fB[n])[1] = *(const uint4*)(srcB[n] + kb + 8);   // next 8 bf16
    }
#pragma unroll
    for (int t = 0; t < MT; t++) {
      const unsigned short* wr = wrow + (size_t)t * 16 * KD;
      v16bf fA;
      ((uint4*)&fA)[0] = *(const uint4*)(wr + kb);        // K kb..kb+7   (+hi 16)
      ((uint4*)&fA)[1] = *(const uint4*)(wr + kb + 8);    // K kb+8..15  (+hi 16)
#pragma unroll
      for (int n = 0; n < NT; n++)
        acc[t][n] = __builtin_amdgcn_wmma_f32_16x16x32_bf16(
                      false, fA, false, fB[n], (short)0, acc[t][n], false, false);
    }
  }

  // ---- phase 3: bias (+ReLU) + store.  C/D layout: lane=col, vgpr r=row(+8 hi)
  const int rowOff = hi ? 8 : 0;
#pragma unroll
  for (int t = 0; t < MT; t++) {
#pragma unroll
    for (int n = 0; n < NT; n++) {
      int pxg = pxBase + n * 16 + ln16;
#pragma unroll
      for (int r = 0; r < 8; r++) {
        int o = mBase + t * 16 + rowOff + r;
        if (MODE == 1) {
          float vv = acc[t][n][r] + bias[o];
          vv = fmaxf(vv, 0.f);
          out[((size_t)(b * Cc + o)) * HW + pos0 + pxg] = vv;
        } else if (o < 27) {
          out[((size_t)(b * 27 + o)) * HW + pos0 + pxg] = acc[t][n][r] + bias[o];
        }
      }
    }
  }
}

// ============================================================
extern "C" void kernel_launch(void* const* d_in, const int* in_sizes, int n_in,
                              void* d_out, int out_size, void* d_ws, size_t ws_size,
                              hipStream_t stream) {
  const float* x = (const float*)d_in[0];
  char* ws = (char*)d_ws;
  size_t off = 0;
  float* xbuf0 = (float*)(ws + off); off += (size_t)Bb * Cc * HW * 4;
  float* xbuf1 = (float*)(ws + off); off += (size_t)Bb * Cc * HW * 4;
  float* om    = (float*)(ws + off); off += (size_t)Bb * 27 * HW * 4;
  unsigned short *wm[3], *wo[3];
  for (int l = 0; l < 3; l++) { wm[l] = (unsigned short*)(ws + off); off += (size_t)256 * KD * 2; }
  for (int l = 0; l < 3; l++) { wo[l] = (unsigned short*)(ws + off); off += (size_t)64  * KD * 2; }

  // allow >64KB dynamic LDS (CDNA5 WGP has 320KB)
  {
    auto kOff  = deform_gemm<0, 1, 1>;
    auto kMain = deform_gemm<1, 2, 2>;
    hipFuncSetAttribute(reinterpret_cast<const void*>(kOff),
                        hipFuncAttributeMaxDynamicSharedMemorySize, SMEM_BYTES);
    hipFuncSetAttribute(reinterpret_cast<const void*>(kMain),
                        hipFuncAttributeMaxDynamicSharedMemorySize, SMEM_BYTES);
  }

  for (int l = 0; l < 3; l++) {
    const float* w_off = (const float*)d_in[1 + 4 * l + 0];
    const float* b_off = (const float*)d_in[1 + 4 * l + 1];
    const float* wmain = (const float*)d_in[1 + 4 * l + 2];
    const float* bmain = (const float*)d_in[1 + 4 * l + 3];

    cvt_w_off <<<(64  * KD + 255) / 256, 256, 0, stream>>>(w_off, wo[l]);
    cvt_w_main<<<(256 * KD + 255) / 256, 256, 0, stream>>>(wmain, wm[l]);

    const float* xin  = (l == 0) ? x : ((l == 1) ? xbuf0 : xbuf1);
    float*       xout = (l == 2) ? (float*)d_out : ((l == 0) ? xbuf0 : xbuf1);

    // offset conv (im2col GEMM, M=64 incl. padding) -> om
    deform_gemm<0, 1, 1><<<NPIX / PX, 256, SMEM_BYTES, stream>>>(xin, nullptr, wo[l], b_off, om);
    // deformable conv GEMM (M=256) -> relu output
    deform_gemm<1, 2, 2><<<NPIX / PX, 256, SMEM_BYTES, stream>>>(xin, om, wm[l], bmain, xout);
  }
}
